// AttentionDecoder_66864050864601
// MI455X (gfx1250) — matrix operand order repaired and checked by reading, not compiled, not run on hardware
//
#include <hip/hip_runtime.h>
#include <hip/hip_bf16.h>
#include <math.h>

// Problem sizes (fixed by the reference)
#define Hh   256
#define Ee   128
#define VOC  50000
#define SEQ  400
#define Bb   256
#define EXTN 100
#define VEXT 50100   // VOC + EXT

// ---------------------------------------------------------------------------
// WMMA bf16 16x16x32 fragment helpers (wave32, CDNA5).
// Layouts per CDNA5 ISA 7.12.2:
//   A (16x32, 16-bit): lane = m + 16*halfK ; VGPR v, pos p:
//       k = (v<4 ? 2v+p : 16+2(v-4)+p) + 8*halfK
//   B (32x16, 16-bit): lane = n + 16*halfK ; k = (v<4 ? 2v+p : 16+2(v-4)+p) + 8*halfK
//   C/D (16x16 f32):   lane = n + 16*halfM ; VGPR r -> m = r + 8*halfM
// ---------------------------------------------------------------------------
typedef __attribute__((ext_vector_type(16))) __bf16 v16bf;
typedef __attribute__((ext_vector_type(8)))  float  v8f;

__device__ __forceinline__ int a_kidx(int e, int half) {
  int v = e >> 1, p = e & 1;
  int k = (v < 4) ? (2 * v + p) : (16 + 2 * (v - 4) + p);
  return k + half * 8;
}
__device__ __forceinline__ int b_kidx(int e, int half) {
  int v = e >> 1, p = e & 1;
  return ((v < 4) ? (2 * v + p) : (16 + 2 * (v - 4) + p)) + half * 8;
}

template <typename F>
__device__ __forceinline__ v16bf load_a_frag(F fetch, int lane) {
  int m = lane & 15, half = lane >> 4;
  v16bf a;
#pragma unroll
  for (int e = 0; e < 16; ++e) a[e] = (__bf16)fetch(m, a_kidx(e, half));
  return a;
}
template <typename F>
__device__ __forceinline__ v16bf load_b_frag(F fetch, int lane) {
  int n = lane & 15, half = lane >> 4;
  v16bf b;
#pragma unroll
  for (int e = 0; e < 16; ++e) b[e] = (__bf16)fetch(b_kidx(e, half), n);
  return b;
}
template <typename G>
__device__ __forceinline__ void store_c_frag(const v8f& c, G put, int lane) {
  int n = lane & 15, mh = (lane >> 4) * 8;
#pragma unroll
  for (int r = 0; r < 8; ++r) put(mh + r, n, c[r]);
}

__device__ __forceinline__ float sigf(float x) { return 1.f / (1.f + expf(-x)); }

// ---------------------------------------------------------------------------
// K1a: embedding gather + x = [emb, c_t_1] @ xctx_w + xctx_b   (B x 128)
// ---------------------------------------------------------------------------
__global__ void k_front(const int* __restrict__ y, const float* __restrict__ c_t1,
                        const float* __restrict__ emb_table,
                        const float* __restrict__ xw, const float* __restrict__ xb,
                        float* __restrict__ X, float* __restrict__ EMB) {
  int b = blockIdx.x, j = threadIdx.x;          // 128 threads
  int tok = y[b];
  const float* er = emb_table + (size_t)tok * Ee;
  EMB[b * Ee + j] = er[j];
  float acc = xb[j];
  for (int k = 0; k < Ee; ++k)       acc += er[k] * xw[k * Ee + j];
  const float* cr = c_t1 + (size_t)b * (2 * Hh);
  for (int k = 0; k < 2 * Hh; ++k)   acc += cr[k] * xw[(Ee + k) * Ee + j];
  X[b * Ee + j] = acc;
}

// ---------------------------------------------------------------------------
// K1b: gates = [x | h_prev] @ [Wih ; Whh] + b_lstm   (256 x 1024, K=384)
// WMMA bf16: 64 N-tile blocks, 8 waves/block, 2 M-tiles/wave, 12 k-steps
// ---------------------------------------------------------------------------
__global__ void k_gates_wmma(const float* __restrict__ X, const float* __restrict__ h_prev,
                             const float* __restrict__ Wih, const float* __restrict__ Whh,
                             const float* __restrict__ b_lstm, float* __restrict__ gates) {
  int lane = threadIdx.x & 31, w = threadIdx.x >> 5;
  int n0 = blockIdx.x * 16;
  int m0a = w * 16, m0b = (w + 8) * 16;
  v8f acc0 = {}, acc1 = {};
  for (int kt = 0; kt < 12; ++kt) {
    int k0 = kt * 32;
    v16bf bf = load_b_frag([&](int k, int n) {
      int kk = k0 + k;
      return (kk < Ee) ? Wih[(size_t)kk * (4 * Hh) + n0 + n]
                       : Whh[(size_t)(kk - Ee) * (4 * Hh) + n0 + n];
    }, lane);
    v16bf a0 = load_a_frag([&](int m, int k) {
      int kk = k0 + k, row = m0a + m;
      return (kk < Ee) ? X[row * Ee + kk] : h_prev[row * Hh + (kk - Ee)];
    }, lane);
    v16bf a1 = load_a_frag([&](int m, int k) {
      int kk = k0 + k, row = m0b + m;
      return (kk < Ee) ? X[row * Ee + kk] : h_prev[row * Hh + (kk - Ee)];
    }, lane);
    acc0 = __builtin_amdgcn_wmma_f32_16x16x32_bf16(false, a0, false, bf, (short)0, acc0, false, false);
    acc1 = __builtin_amdgcn_wmma_f32_16x16x32_bf16(false, a1, false, bf, (short)0, acc1, false, false);
  }
  store_c_frag(acc0, [&](int m, int n, float v) {
    gates[(size_t)(m0a + m) * (4 * Hh) + n0 + n] = v + b_lstm[n0 + n]; }, lane);
  store_c_frag(acc1, [&](int m, int n, float v) {
    gates[(size_t)(m0b + m) * (4 * Hh) + n0 + n] = v + b_lstm[n0 + n]; }, lane);
}

// ---------------------------------------------------------------------------
// K1c: LSTM cell elementwise -> h_new, c_new (written into d_out)
// ---------------------------------------------------------------------------
__global__ void k_lstm(const float* __restrict__ gates, const float* __restrict__ c_prev,
                       float* __restrict__ h_out, float* __restrict__ c_out) {
  int b = blockIdx.x, h = threadIdx.x;          // 256 threads
  const float* g = gates + (size_t)b * (4 * Hh);
  float ig = g[h], fg = g[Hh + h], gg = g[2 * Hh + h], og = g[3 * Hh + h];
  float cn = sigf(fg) * c_prev[b * Hh + h] + sigf(ig) * tanhf(gg);
  float hn = sigf(og) * tanhf(cn);
  c_out[b * Hh + h] = cn;
  h_out[b * Hh + h] = hn;
}

// ---------------------------------------------------------------------------
// K1d: s_proj = [h_new | c_new] @ Ws_w + Ws_b   (256 x 512, K=512) via WMMA
// ---------------------------------------------------------------------------
__global__ void k_sproj_wmma(const float* __restrict__ h_new, const float* __restrict__ c_new,
                             const float* __restrict__ Ws_w, const float* __restrict__ Ws_b,
                             float* __restrict__ sproj) {
  int lane = threadIdx.x & 31, w = threadIdx.x >> 5;
  int n0 = blockIdx.x * 16;                     // 32 blocks
  int m0a = w * 16, m0b = (w + 8) * 16;
  v8f acc0 = {}, acc1 = {};
  for (int kt = 0; kt < 16; ++kt) {
    int k0 = kt * 32;
    v16bf bf = load_b_frag([&](int k, int n) {
      return Ws_w[(size_t)(k0 + k) * (2 * Hh) + n0 + n]; }, lane);
    v16bf a0 = load_a_frag([&](int m, int k) {
      int kk = k0 + k, row = m0a + m;
      return (kk < Hh) ? h_new[row * Hh + kk] : c_new[row * Hh + (kk - Hh)];
    }, lane);
    v16bf a1 = load_a_frag([&](int m, int k) {
      int kk = k0 + k, row = m0b + m;
      return (kk < Hh) ? h_new[row * Hh + kk] : c_new[row * Hh + (kk - Hh)];
    }, lane);
    acc0 = __builtin_amdgcn_wmma_f32_16x16x32_bf16(false, a0, false, bf, (short)0, acc0, false, false);
    acc1 = __builtin_amdgcn_wmma_f32_16x16x32_bf16(false, a1, false, bf, (short)0, acc1, false, false);
  }
  store_c_frag(acc0, [&](int m, int n, float v) {
    sproj[(size_t)(m0a + m) * (2 * Hh) + n0 + n] = v + Ws_b[n0 + n]; }, lane);
  store_c_frag(acc1, [&](int m, int n, float v) {
    sproj[(size_t)(m0b + m) * (2 * Hh) + n0 + n] = v + Ws_b[n0 + n]; }, lane);
}

// ---------------------------------------------------------------------------
// K2: e_t[b,s] = sum_d tanh(enc_features + s_proj + cov*Wcov) * v_w
// One wave per (b,s); 512-wide reduction, float4 streaming (210 MB read)
// ---------------------------------------------------------------------------
__global__ void k_scores(const float* __restrict__ ef, const float* __restrict__ sproj,
                         const float* __restrict__ cov, const float* __restrict__ vw,
                         const float* __restrict__ wcov, float* __restrict__ et) {
  int lane = threadIdx.x & 31, w = threadIdx.x >> 5;
  int p = blockIdx.x * 8 + w;                   // 102400 pairs
  int b = p / SEQ, s = p % SEQ;
  const float* efp = ef + ((size_t)b * SEQ + s) * (2 * Hh);
  const float* spp = sproj + (size_t)b * (2 * Hh);
  float cv = cov[p];
  float sum = 0.f;
#pragma unroll
  for (int i = 0; i < 4; ++i) {
    int d = i * 128 + lane * 4;
    float4 e  = *(const float4*)(efp + d);
    float4 sp = *(const float4*)(spp + d);
    float4 v  = *(const float4*)(vw + d);
    float4 wc = *(const float4*)(wcov + d);
    sum += tanhf(e.x + sp.x + cv * wc.x) * v.x;
    sum += tanhf(e.y + sp.y + cv * wc.y) * v.y;
    sum += tanhf(e.z + sp.z + cv * wc.z) * v.z;
    sum += tanhf(e.w + sp.w + cv * wc.w) * v.w;
  }
#pragma unroll
  for (int off = 16; off > 0; off >>= 1) sum += __shfl_xor(sum, off);
  if (lane == 0) et[p] = sum;
}

// ---------------------------------------------------------------------------
// K3: masked softmax over S + renorm + coverage update (block per b)
// ---------------------------------------------------------------------------
__global__ void k_softmax_attn(const float* __restrict__ et, const float* __restrict__ mask,
                               const float* __restrict__ cov, float* __restrict__ attn_out,
                               float* __restrict__ cov_out) {
  __shared__ float red[256];
  int b = blockIdx.x, t = threadIdx.x;
  const float* e = et + (size_t)b * SEQ;
  float m = -INFINITY;
  for (int s = t; s < SEQ; s += 256) m = fmaxf(m, e[s]);
  red[t] = m; __syncthreads();
  for (int o = 128; o; o >>= 1) { if (t < o) red[t] = fmaxf(red[t], red[t + o]); __syncthreads(); }
  m = red[0]; __syncthreads();
  float z = 0.f;
  for (int s = t; s < SEQ; s += 256) z += expf(e[s] - m);
  red[t] = z; __syncthreads();
  for (int o = 128; o; o >>= 1) { if (t < o) red[t] += red[t + o]; __syncthreads(); }
  float Z = red[0]; __syncthreads();
  float r = 0.f;
  for (int s = t; s < SEQ; s += 256) r += expf(e[s] - m) * mask[b * SEQ + s] / Z;
  red[t] = r; __syncthreads();
  for (int o = 128; o; o >>= 1) { if (t < o) red[t] += red[t + o]; __syncthreads(); }
  float R = red[0] + 1e-12f;
  for (int s = t; s < SEQ; s += 256) {
    float a = expf(e[s] - m) * mask[b * SEQ + s] / Z;
    float ad = a / R;
    attn_out[b * SEQ + s] = ad;
    cov_out[b * SEQ + s]  = cov[b * SEQ + s] + ad;
  }
}

// ---------------------------------------------------------------------------
// K4: ctx[b,d] = sum_s attn[b,s] * enc_outputs[b,s,d]  (210 MB stream)
// ---------------------------------------------------------------------------
__global__ void k_ctx(const float* __restrict__ attn, const float* __restrict__ eo,
                      float* __restrict__ ctx) {
  __shared__ float a[SEQ];
  int b = blockIdx.x, t = threadIdx.x;          // 512 threads
  for (int s = t; s < SEQ; s += 512) a[s] = attn[b * SEQ + s];
  __syncthreads();
  const float* p = eo + (size_t)b * SEQ * (2 * Hh) + t;
  float acc = 0.f;
  for (int s = 0; s < SEQ; ++s) {
    if (s + 8 < SEQ) __builtin_prefetch(p + (size_t)(s + 8) * (2 * Hh), 0, 0);
    acc += a[s] * p[(size_t)s * (2 * Hh)];
  }
  ctx[b * (2 * Hh) + t] = acc;
}

// ---------------------------------------------------------------------------
// K5: dec = [h_new | ctx] @ V1_w + V1_b ; p_gen = sigma([ctx,s_cat,emb].pgen_w + b)
// ---------------------------------------------------------------------------
__global__ void k_dec_pgen(const float* __restrict__ h_new, const float* __restrict__ c_new,
                           const float* __restrict__ ctx, const float* __restrict__ emb,
                           const float* __restrict__ V1w, const float* __restrict__ V1b,
                           const float* __restrict__ pgw, const float* __restrict__ pgb,
                           float* __restrict__ dec, float* __restrict__ pgen) {
  __shared__ float s2[3 * Hh];
  __shared__ float red[256];
  int b = blockIdx.x, t = threadIdx.x;          // 256 threads
  s2[t] = h_new[b * Hh + t];
  s2[Hh + t]      = ctx[b * (2 * Hh) + t];
  s2[Hh + 256 + t] = ctx[b * (2 * Hh) + 256 + t];
  __syncthreads();
  float acc = V1b[t];
  for (int k = 0; k < 3 * Hh; ++k) acc += s2[k] * V1w[(size_t)k * Hh + t];
  dec[b * Hh + t] = acc;
  // p_gen: z = [ctx(512) | h_new(256) | c_new(256) | emb(128)] . pgen_w (1152)
  float ps = 0.f;
  for (int k = t; k < 4 * Hh + Ee; k += 256) {
    float zv;
    if (k < 2 * Hh)           zv = ctx[b * (2 * Hh) + k];
    else if (k < 3 * Hh)      zv = h_new[b * Hh + (k - 2 * Hh)];
    else if (k < 4 * Hh)      zv = c_new[b * Hh + (k - 3 * Hh)];
    else                      zv = emb[b * Ee + (k - 4 * Hh)];
    ps += zv * pgw[k];
  }
  red[t] = ps; __syncthreads();
  for (int o = 128; o; o >>= 1) { if (t < o) red[t] += red[t + o]; __syncthreads(); }
  if (t == 0) pgen[b] = sigf(red[0] + pgb[0]);
}

// ---------------------------------------------------------------------------
// K6: vocab logits = dec @ V2_w + V2_b  (256 x 50000, K=256), WMMA bf16.
// Logits staged directly in the final_dist region of d_out (row stride 50100).
// 3125 N-tile blocks; each block covers full M=256 so V2_w is read once.
// ---------------------------------------------------------------------------
__global__ void k_vocab_wmma(const float* __restrict__ dec, const float* __restrict__ V2w,
                             const float* __restrict__ V2b, float* __restrict__ outF) {
  int lane = threadIdx.x & 31, w = threadIdx.x >> 5;
  int n0 = blockIdx.x * 16;                     // 3125 blocks
  int m0a = w * 16, m0b = (w + 8) * 16;
  v8f acc0 = {}, acc1 = {};
  for (int kt = 0; kt < 8; ++kt) {
    int k0 = kt * 32;
    v16bf bf = load_b_frag([&](int k, int n) {
      return V2w[(size_t)(k0 + k) * VOC + n0 + n]; }, lane);
    v16bf a0 = load_a_frag([&](int m, int k) {
      return dec[(size_t)(m0a + m) * Hh + k0 + k]; }, lane);
    v16bf a1 = load_a_frag([&](int m, int k) {
      return dec[(size_t)(m0b + m) * Hh + k0 + k]; }, lane);
    acc0 = __builtin_amdgcn_wmma_f32_16x16x32_bf16(false, a0, false, bf, (short)0, acc0, false, false);
    acc1 = __builtin_amdgcn_wmma_f32_16x16x32_bf16(false, a1, false, bf, (short)0, acc1, false, false);
  }
  store_c_frag(acc0, [&](int m, int n, float v) {
    outF[(size_t)(m0a + m) * VEXT + n0 + n] = v + V2b[n0 + n]; }, lane);
  store_c_frag(acc1, [&](int m, int n, float v) {
    outF[(size_t)(m0b + m) * VEXT + n0 + n] = v + V2b[n0 + n]; }, lane);
}

// ---------------------------------------------------------------------------
// K7: per-row online softmax stats (max, sumexp) over 50000 logits
// ---------------------------------------------------------------------------
__global__ void k_rowstats(const float* __restrict__ outF, float* __restrict__ rstat) {
  __shared__ float rm[256], rs[256];
  int b = blockIdx.x, t = threadIdx.x;
  const float* row = outF + (size_t)b * VEXT;
  float m = -INFINITY, s = 0.f;
  for (int v = t; v < VOC; v += 256) {
    float x = row[v];
    if (x > m) { s = s * expf(m - x) + 1.f; m = x; }
    else       { s += expf(x - m); }
  }
  rm[t] = m; rs[t] = s; __syncthreads();
  for (int o = 128; o; o >>= 1) {
    if (t < o) {
      float M = fmaxf(rm[t], rm[t + o]);
      rs[t] = rs[t] * expf(rm[t] - M) + rs[t + o] * expf(rm[t + o] - M);
      rm[t] = M;
    }
    __syncthreads();
  }
  if (t == 0) { rstat[2 * b] = rm[0]; rstat[2 * b + 1] = rs[0]; }
}

// ---------------------------------------------------------------------------
// K8: final_dist = p_gen * softmax(logits) (in place); EXT tail zeroed
// ---------------------------------------------------------------------------
__global__ void k_finalize(float* __restrict__ outF, const float* __restrict__ rstat,
                           const float* __restrict__ pgen) {
  size_t i = (size_t)blockIdx.x * 256 + threadIdx.x;
  if (i >= (size_t)Bb * VEXT) return;
  int b = (int)(i / VEXT);
  int v = (int)(i % VEXT);
  if (v < VOC) {
    float m = rstat[2 * b], Z = rstat[2 * b + 1];
    outF[i] = pgen[b] * expf(outF[i] - m) / Z;
  } else {
    outF[i] = 0.f;
  }
}

// ---------------------------------------------------------------------------
// K9: scatter-add copy distribution: out[b, idx[b,s]] += (1-p_gen)*attn[b,s]
// ---------------------------------------------------------------------------
__global__ void k_scatter(float* __restrict__ outF, const int* __restrict__ idx,
                          const float* __restrict__ attn, const float* __restrict__ pgen) {
  int i = blockIdx.x * 256 + threadIdx.x;       // 102400
  int b = i / SEQ;
  float wgt = (1.f - pgen[b]) * attn[i];
  atomicAdd(outF + (size_t)b * VEXT + idx[i], wgt);
}

// ---------------------------------------------------------------------------
// Host orchestration
// ---------------------------------------------------------------------------
extern "C" void kernel_launch(void* const* d_in, const int* in_sizes, int n_in,
                              void* d_out, int out_size, void* d_ws, size_t ws_size,
                              hipStream_t stream) {
  const int*   y_t_1   = (const int*)  d_in[0];
  const float* h_prev  = (const float*)d_in[1];
  const float* c_prev  = (const float*)d_in[2];
  const float* c_t_1   = (const float*)d_in[3];
  const float* enc_out = (const float*)d_in[4];
  const float* enc_ftr = (const float*)d_in[5];
  const float* enc_msk = (const float*)d_in[6];
  const int*   enc_idx = (const int*)  d_in[8];
  const float* cov_t   = (const float*)d_in[9];
  const float* emb_tab = (const float*)d_in[10];
  const float* Ws_w    = (const float*)d_in[11];
  const float* Ws_b    = (const float*)d_in[12];
  const float* v_w     = (const float*)d_in[13];
  const float* xctx_w  = (const float*)d_in[14];
  const float* xctx_b  = (const float*)d_in[15];
  const float* Wih     = (const float*)d_in[16];
  const float* Whh     = (const float*)d_in[17];
  const float* b_lstm  = (const float*)d_in[18];
  const float* V1_w    = (const float*)d_in[19];
  const float* V1_b    = (const float*)d_in[20];
  const float* V2_w    = (const float*)d_in[21];
  const float* V2_b    = (const float*)d_in[22];
  const float* pgen_w  = (const float*)d_in[23];
  const float* pgen_b  = (const float*)d_in[24];
  const float* Wcov_w  = (const float*)d_in[25];

  float* out = (float*)d_out;
  // Output layout (floats): final(256*50100) | h(65536) | c(65536) | ctx(131072) | attn(102400) | cov(102400)
  float* out_final = out;
  float* out_h     = out + (size_t)Bb * VEXT;
  float* out_c     = out_h + (size_t)Bb * Hh;
  float* out_ctx   = out_c + (size_t)Bb * Hh;
  float* out_attn  = out_ctx + (size_t)Bb * 2 * Hh;
  float* out_cov   = out_attn + (size_t)Bb * SEQ;

  // Workspace layout (floats)
  float* ws = (float*)d_ws;
  float* wX     = ws;                       // 256*128
  float* wEMB   = wX    + Bb * Ee;          // 256*128
  float* wGATES = wEMB  + Bb * Ee;          // 256*1024
  float* wSPROJ = wGATES + Bb * 4 * Hh;     // 256*512
  float* wET    = wSPROJ + Bb * 2 * Hh;     // 256*400
  float* wDEC   = wET   + Bb * SEQ;         // 256*256
  float* wPGEN  = wDEC  + Bb * Hh;          // 256
  float* wRSTAT = wPGEN + Bb;               // 512

  // 1) embedding + xctx
  k_front<<<Bb, Ee, 0, stream>>>(y_t_1, c_t_1, emb_tab, xctx_w, xctx_b, wX, wEMB);
  // 2) LSTM gates GEMM (WMMA bf16)
  k_gates_wmma<<<(4 * Hh) / 16, 256, 0, stream>>>(wX, h_prev, Wih, Whh, b_lstm, wGATES);
  // 3) LSTM cell
  k_lstm<<<Bb, Hh, 0, stream>>>(wGATES, c_prev, out_h, out_c);
  // 4) state projection GEMM (WMMA bf16)
  k_sproj_wmma<<<(2 * Hh) / 16, 256, 0, stream>>>(out_h, out_c, Ws_w, Ws_b, wSPROJ);
  // 5) attention scores (streams enc_features)
  k_scores<<<(Bb * SEQ) / 8, 256, 0, stream>>>(enc_ftr, wSPROJ, cov_t, v_w, Wcov_w, wET);
  // 6) masked softmax + coverage
  k_softmax_attn<<<Bb, 256, 0, stream>>>(wET, enc_msk, cov_t, out_attn, out_cov);
  // 7) context vector (streams enc_outputs)
  k_ctx<<<Bb, 2 * Hh, 0, stream>>>(out_attn, enc_out, out_ctx);
  // 8) dec projection + p_gen
  k_dec_pgen<<<Bb, Hh, 0, stream>>>(out_h, out_c, out_ctx, wEMB, V1_w, V1_b,
                                    pgen_w, pgen_b, wDEC, wPGEN);
  // 9) vocab GEMM (WMMA bf16) -> logits staged in out_final
  k_vocab_wmma<<<VOC / 16, 256, 0, stream>>>(wDEC, V2_w, V2_b, out_final);
  // 10) online softmax stats per row
  k_rowstats<<<Bb, 256, 0, stream>>>(out_final, wRSTAT);
  // 11) in-place softmax * p_gen, zero EXT tail
  {
    size_t total = (size_t)Bb * VEXT;
    k_finalize<<<(unsigned)((total + 255) / 256), 256, 0, stream>>>(out_final, wRSTAT, wPGEN);
  }
  // 12) pointer scatter-add
  k_scatter<<<(Bb * SEQ) / 256, 256, 0, stream>>>(out_final, enc_idx, out_attn, wPGEN);
}
